// GlobalContextBlock_68624987455943
// MI455X (gfx1250) — compile-verified, matrix operand order checked by reading, not run on hardware
//
#include <hip/hip_runtime.h>
#include <hip/hip_bf16.h>

// ---------------------------------------------------------------------------
// Problem sizes (fixed by the reference)
// ---------------------------------------------------------------------------
#define BB   4
#define LL   2048
#define DD   768
#define NN   16
#define MM   (BB * LL)          // 8192 rows of x
#define KT   (DD / 32)          // 24 K-tiles of 32
#define TM   (MM / 16)          // 512 M-tiles
#define TN_DT (DD / 16)         // 48 N-tiles for the dt GEMM
#define CH   8                  // scan chunks
#define LC   (LL / CH)          // 256 steps per chunk
#define LN_EPS 1e-5f

typedef __bf16 bf16_t;
typedef bf16_t v16bf __attribute__((ext_vector_type(16)));
typedef float  v8f   __attribute__((ext_vector_type(8)));

union FragU { uint4 q[2]; v16bf v; };

__device__ __forceinline__ unsigned bf16_rne(float f) {
    unsigned u = __float_as_uint(f);
    return (u + 0x7FFFu + ((u >> 16) & 1u)) >> 16;
}
__device__ __forceinline__ unsigned pack_bf16x2(float lo, float hi) {
    return bf16_rne(lo) | (bf16_rne(hi) << 16);
}

// ---------------------------------------------------------------------------
// Pack x (M x K f32, row major) into WMMA A-fragment layout, bf16.
// Tile = 16M x 32K. Per lane: 8 dwords (16 bf16) exactly as ISA 7.12.2:
//   lane<16: M=lane,   K = {2j,2j+1}        (j<4)  / {16+2(j-4),..} (j>=4)
//   lane>=16: same M rows, K pattern +8
// ---------------------------------------------------------------------------
__global__ __launch_bounds__(256)
void pack_a_kernel(const float* __restrict__ x, unsigned* __restrict__ apk) {
    int idx  = blockIdx.x * 256 + threadIdx.x;     // TM*KT*256 threads
    int j    = idx & 7;
    int lane = (idx >> 3) & 31;
    int tile = idx >> 8;
    int tk = tile % KT;
    int tm = tile / KT;
    int m  = tm * 16 + (lane & 15);
    int kin = ((j < 4) ? (2 * j) : (16 + 2 * (j - 4))) + ((lane >> 4) << 3);
    int k  = tk * 32 + kin;
    const float* p = x + (size_t)m * DD + k;
    apk[(size_t)tile * 256 + lane * 8 + j] = pack_bf16x2(p[0], p[1]);
}

// ---------------------------------------------------------------------------
// Pack a weight matrix W (K x Ncols f32) into B-fragment layout (32K x 16N
// tiles), lane = column, same K striping as A, two consecutive K per dword.
// ---------------------------------------------------------------------------
__global__ __launch_bounds__(256)
void pack_b_kernel(const float* __restrict__ w, unsigned* __restrict__ bpk,
                   int ncols) {
    int idx  = blockIdx.x * 256 + threadIdx.x;     // numTileN*KT*256 threads
    int j    = idx & 7;
    int lane = (idx >> 3) & 31;
    int tile = idx >> 8;
    int tk = tile % KT;
    int tn = tile / KT;
    int n  = tn * 16 + (lane & 15);
    int kin = ((j < 4) ? (2 * j) : (16 + 2 * (j - 4))) + ((lane >> 4) << 3);
    int k  = tk * 32 + kin;
    float a = w[(size_t)k * ncols + n];
    float b = w[(size_t)(k + 1) * ncols + n];
    bpk[(size_t)tile * 256 + lane * 8 + j] = pack_bf16x2(a, b);
}

// ---------------------------------------------------------------------------
// WMMA GEMM: out(M x Ncols) = A(M x 768) * B(768 x Ncols) [+ bias, softplus]
// One wave owns a 16M x (16*NT) output strip. Software-pipelined: all A/B
// fragment loads for K-step tk+1 are issued before the WMMAs of step tk, so
// s_wait_loadcnt becomes partial and v_wmma can issue back-to-back.
// MODE 0: plain store.  MODE 1: v = softplus(v + bias[n]).
// ---------------------------------------------------------------------------
template <int NT, int MODE>
__global__ __launch_bounds__(128)
void gemm_wmma_kernel(const uint4* __restrict__ Ap, const uint4* __restrict__ Bp,
                      const float* __restrict__ bias, float* __restrict__ out,
                      int ncols, int numTileN) {
    int wave = threadIdx.x >> 5;
    int lane = threadIdx.x & 31;
    int tm   = blockIdx.x;
    int tn0  = (blockIdx.y * (blockDim.x >> 5) + wave) * NT;
    if (tn0 >= numTileN) return;

    v8f acc[NT];
    v8f zero = {0.f, 0.f, 0.f, 0.f, 0.f, 0.f, 0.f, 0.f};
#pragma unroll
    for (int t = 0; t < NT; ++t) acc[t] = zero;

    const uint4* apBase = Ap + (size_t)tm * KT * 64 + lane * 2;
    const uint4* bpBase = Bp + (size_t)tn0 * KT * 64 + lane * 2;

    FragU fa[2];
    FragU fb[2][NT];

    // prologue: stage tk = 0
    fa[0].q[0] = apBase[0];
    fa[0].q[1] = apBase[1];
#pragma unroll
    for (int t = 0; t < NT; ++t) {
        const uint4* bp = bpBase + (size_t)t * KT * 64;
        fb[0][t].q[0] = bp[0];
        fb[0][t].q[1] = bp[1];
    }

#pragma unroll 2
    for (int tk = 0; tk < KT; ++tk) {
        int cur = tk & 1;
        int nxt = cur ^ 1;
        if (tk + 1 < KT) {
            // stage tk+1 while tk's WMMAs run
            fa[nxt].q[0] = apBase[(size_t)(tk + 1) * 64 + 0];
            fa[nxt].q[1] = apBase[(size_t)(tk + 1) * 64 + 1];
#pragma unroll
            for (int t = 0; t < NT; ++t) {
                const uint4* bp = bpBase + ((size_t)t * KT + tk + 1) * 64;
                fb[nxt][t].q[0] = bp[0];
                fb[nxt][t].q[1] = bp[1];
            }
        }
        if (tk + 2 < KT)  // WGP-scope prefetch of the A tile after next
            __builtin_prefetch(apBase + (size_t)(tk + 2) * 64, 0, 3);
#pragma unroll
        for (int t = 0; t < NT; ++t) {
            acc[t] = __builtin_amdgcn_wmma_f32_16x16x32_bf16(
                false, fa[cur].v, false, fb[cur][t].v, (short)0, acc[t],
                false, false);
        }
    }

    // C/D layout: lane<16 rows 0..7 in acc[0..7]; lane>=16 rows 8..15.
    int col   = lane & 15;
    int rbase = (lane >> 4) * 8;
#pragma unroll
    for (int t = 0; t < NT; ++t) {
        int n = (tn0 + t) * 16 + col;
        float bv = (MODE == 1) ? bias[n] : 0.f;
#pragma unroll
        for (int r = 0; r < 8; ++r) {
            int m = tm * 16 + rbase + r;
            float v = acc[t][r] + bv;
            if (MODE == 1) v = (v > 20.f) ? v : logf(1.f + expf(v));
            out[(size_t)m * ncols + n] = v;
        }
    }
}

// ---------------------------------------------------------------------------
// Chunked scan, pass A: per (b,d,n,chunk) compute local scan with h0=0 and
// the running product P = prod(dA) over the chunk. Lane = n + 16*(d sub).
// ---------------------------------------------------------------------------
__global__ __launch_bounds__(256)
void scan_chunks_kernel(const float* __restrict__ x, const float* __restrict__ dt,
                        const float* __restrict__ Bm, const float* __restrict__ A_log,
                        float* __restrict__ Pbuf, float* __restrict__ Hloc) {
    int tid = threadIdx.x;
    int n   = tid & 15;
    int doff = tid >> 4;
    int blk = blockIdx.x;                 // B * (D/16) * CH
    int chunk = blk % CH;
    int dblk  = (blk / CH) % (DD / 16);
    int b     = blk / (CH * (DD / 16));
    int d = dblk * 16 + doff;

    float a = -expf(A_log[d * NN + n]);
    float h = 0.f, P = 1.f;
    const float* dtp = dt + (size_t)b * LL * DD + d;
    const float* xp  = x  + (size_t)b * LL * DD + d;
    const float* bp  = Bm + (size_t)b * LL * NN + n;
    int l0 = chunk * LC;
    for (int l = l0; l < l0 + LC; ++l) {
        float dtv = dtp[(size_t)l * DD];
        float xv  = xp[(size_t)l * DD];
        float Bv  = bp[(size_t)l * NN];
        float dA  = expf(dtv * a);
        h = dA * h + (dtv * xv) * Bv;
        P *= dA;
    }
    size_t idx = ((size_t)b * DD + d) * NN + n;
    Pbuf[idx * CH + chunk] = P;
    Hloc[idx * CH + chunk] = h;
}

// Pass B: sequential carry over the CH chunk boundaries (tiny).
__global__ __launch_bounds__(256)
void scan_carry_kernel(const float* __restrict__ Pbuf, const float* __restrict__ Hloc,
                       float* __restrict__ Hin) {
    int i = blockIdx.x * 256 + threadIdx.x;    // B*D*N threads
    if (i >= BB * DD * NN) return;
    float h = 0.f;
    for (int s = 0; s < CH; ++s) {
        Hin[(size_t)i * CH + s] = h;
        h = Pbuf[(size_t)i * CH + s] * h + Hloc[(size_t)i * CH + s];
    }
}

// Pass C: re-run each chunk from its corrected h0, emit y = sum_n h*C via a
// width-16 xor-shuffle tree (wave32 halves each own one d channel).
__global__ __launch_bounds__(256)
void scan_emit_kernel(const float* __restrict__ x, const float* __restrict__ dt,
                      const float* __restrict__ Bm, const float* __restrict__ Cm,
                      const float* __restrict__ A_log, const float* __restrict__ Hin,
                      float* __restrict__ y) {
    int tid = threadIdx.x;
    int n   = tid & 15;
    int doff = tid >> 4;
    int blk = blockIdx.x;
    int chunk = blk % CH;
    int dblk  = (blk / CH) % (DD / 16);
    int b     = blk / (CH * (DD / 16));
    int d = dblk * 16 + doff;

    float a = -expf(A_log[d * NN + n]);
    size_t idx = ((size_t)b * DD + d) * NN + n;
    float h = Hin[idx * CH + chunk];

    const float* dtp = dt + (size_t)b * LL * DD + d;
    const float* xp  = x  + (size_t)b * LL * DD + d;
    const float* bp  = Bm + (size_t)b * LL * NN + n;
    const float* cp  = Cm + (size_t)b * LL * NN + n;
    int l0 = chunk * LC;
    for (int l = l0; l < l0 + LC; ++l) {
        float dtv = dtp[(size_t)l * DD];
        float xv  = xp[(size_t)l * DD];
        float Bv  = bp[(size_t)l * NN];
        float Cv  = cp[(size_t)l * NN];
        float dA  = expf(dtv * a);
        h = dA * h + (dtv * xv) * Bv;
        float c = h * Cv;
        c += __shfl_xor(c, 1, 16);
        c += __shfl_xor(c, 2, 16);
        c += __shfl_xor(c, 4, 16);
        c += __shfl_xor(c, 8, 16);
        if (n == 0) y[((size_t)b * LL + l) * DD + d] = c;
    }
}

// ---------------------------------------------------------------------------
// Residual + LayerNorm: r = y + (1 + D_skip)*x ; out = LN(r)*gamma + beta
// One 256-thread block per (b,l) row, 3 elements per thread.
// ---------------------------------------------------------------------------
__global__ __launch_bounds__(256)
void ln_kernel(const float* __restrict__ x, const float* __restrict__ y,
               const float* __restrict__ dsk, const float* __restrict__ gamma,
               const float* __restrict__ beta, float* __restrict__ out) {
    int row = blockIdx.x;          // B*L rows
    int tid = threadIdx.x;
    float r[3];
    float s = 0.f, s2 = 0.f;
#pragma unroll
    for (int i = 0; i < 3; ++i) {
        int d = tid + i * 256;
        float xv = x[(size_t)row * DD + d];
        float yv = y[(size_t)row * DD + d];
        float rv = yv + (1.f + dsk[d]) * xv;
        r[i] = rv;
        s += rv;
        s2 += rv * rv;
    }
#pragma unroll
    for (int off = 16; off; off >>= 1) {
        s  += __shfl_xor(s,  off, 32);
        s2 += __shfl_xor(s2, off, 32);
    }
    __shared__ float ps[8], ps2[8];
    int wave = tid >> 5, lane = tid & 31;
    if (lane == 0) { ps[wave] = s; ps2[wave] = s2; }
    __syncthreads();
    if (wave == 0) {
        float a  = (lane < 8) ? ps[lane]  : 0.f;
        float a2 = (lane < 8) ? ps2[lane] : 0.f;
        for (int off = 4; off; off >>= 1) {
            a  += __shfl_xor(a,  off, 32);
            a2 += __shfl_xor(a2, off, 32);
        }
        if (lane == 0) { ps[0] = a; ps2[0] = a2; }
    }
    __syncthreads();
    float mu   = ps[0] * (1.f / DD);
    float var  = ps2[0] * (1.f / DD) - mu * mu;
    float rstd = rsqrtf(var + LN_EPS);
#pragma unroll
    for (int i = 0; i < 3; ++i) {
        int d = tid + i * 256;
        out[(size_t)row * DD + d] = (r[i] - mu) * rstd * gamma[d] + beta[d];
    }
}

// ---------------------------------------------------------------------------
// Host-side orchestration
// ---------------------------------------------------------------------------
static inline size_t align256(size_t v) { return (v + 255) & ~(size_t)255; }

extern "C" void kernel_launch(void* const* d_in, const int* in_sizes, int n_in,
                              void* d_out, int out_size, void* d_ws, size_t ws_size,
                              hipStream_t stream) {
    const float* x     = (const float*)d_in[0];
    const float* W_dt  = (const float*)d_in[1];
    const float* b_dt  = (const float*)d_in[2];
    const float* W_B   = (const float*)d_in[3];
    const float* W_C   = (const float*)d_in[4];
    const float* A_log = (const float*)d_in[5];
    const float* D_sk  = (const float*)d_in[6];
    const float* gamma = (const float*)d_in[7];
    const float* beta  = (const float*)d_in[8];
    float* out = (float*)d_out;

    // Workspace carve-up (all L2-resident on MI455X: 192 MB L2)
    char* w = (char*)d_ws;
    size_t off = 0;
    unsigned* Apk  = (unsigned*)(w + off); off = align256(off + (size_t)TM * KT * 1024);
    unsigned* Bdt  = (unsigned*)(w + off); off = align256(off + (size_t)TN_DT * KT * 1024);
    unsigned* Bb   = (unsigned*)(w + off); off = align256(off + (size_t)1 * KT * 1024);
    unsigned* Bc   = (unsigned*)(w + off); off = align256(off + (size_t)1 * KT * 1024);
    float* dt_buf  = (float*)(w + off);    off = align256(off + (size_t)MM * DD * 4);
    float* Bm      = (float*)(w + off);    off = align256(off + (size_t)MM * NN * 4);
    float* Cm      = (float*)(w + off);    off = align256(off + (size_t)MM * NN * 4);
    float* Pbuf    = (float*)(w + off);    off = align256(off + (size_t)BB * DD * NN * CH * 4);
    float* Hloc    = (float*)(w + off);    off = align256(off + (size_t)BB * DD * NN * CH * 4);
    float* Hin     = (float*)(w + off);    off = align256(off + (size_t)BB * DD * NN * CH * 4);
    float* y_buf   = (float*)(w + off);    off = align256(off + (size_t)MM * DD * 4);
    (void)ws_size; (void)n_in; (void)in_sizes; (void)out_size;

    // 1) bf16 pack (f32 -> WMMA fragment layout)
    pack_a_kernel<<<(TM * KT * 256) / 256, 256, 0, stream>>>(x, Apk);
    pack_b_kernel<<<(TN_DT * KT * 256) / 256, 256, 0, stream>>>(W_dt, Bdt, DD);
    pack_b_kernel<<<(1 * KT * 256) / 256, 256, 0, stream>>>(W_B, Bb, NN);
    pack_b_kernel<<<(1 * KT * 256) / 256, 256, 0, stream>>>(W_C, Bc, NN);

    // 2) WMMA projections
    //    dt: 48 N-tiles, NT=4 per wave, 4 waves/block -> grid.y = 3
    gemm_wmma_kernel<4, 1><<<dim3(TM, 3), 128, 0, stream>>>(
        (const uint4*)Apk, (const uint4*)Bdt, b_dt, dt_buf, DD, TN_DT);
    //    Bm / Cm: single 16-col tile
    gemm_wmma_kernel<1, 0><<<dim3(TM, 1), 32, 0, stream>>>(
        (const uint4*)Apk, (const uint4*)Bb, nullptr, Bm, NN, 1);
    gemm_wmma_kernel<1, 0><<<dim3(TM, 1), 32, 0, stream>>>(
        (const uint4*)Apk, (const uint4*)Bc, nullptr, Cm, NN, 1);

    // 3) chunked parallel scan (8x parallelism on the latency-bound recurrence)
    int scan_blocks = BB * (DD / 16) * CH;        // 1536 blocks x 256 thr
    scan_chunks_kernel<<<scan_blocks, 256, 0, stream>>>(x, dt_buf, Bm, A_log, Pbuf, Hloc);
    scan_carry_kernel<<<(BB * DD * NN + 255) / 256, 256, 0, stream>>>(Pbuf, Hloc, Hin);
    scan_emit_kernel<<<scan_blocks, 256, 0, stream>>>(x, dt_buf, Bm, Cm, A_log, Hin, y_buf);

    // 4) residual + LayerNorm
    ln_kernel<<<MM, 256, 0, stream>>>(x, y_buf, D_sk, gamma, beta, out);
}
